// TransducerJoint_43662637531360
// MI455X (gfx1250) — matrix operand order repaired and checked by reading, not compiled, not run on hardware
//
#include <hip/hip_runtime.h>

typedef _Float16 half_t;
typedef _Float16 v16h __attribute__((ext_vector_type(16)));
typedef float    v8f  __attribute__((ext_vector_type(8)));

#define ENC_D 256
#define JD    512
#define VOC   1024
#define B_    4
#define T_    256
#define U_    64

__device__ __forceinline__ float fast_tanh(float x) {
  // tanh(x) = sign(x) * (1 - e^{-2|x|}) / (1 + e^{-2|x|}),  e^{-2|x|} = 2^{-2|x|/ln2}
  float ax = __builtin_fabsf(x);
  float e  = __builtin_amdgcn_exp2f(ax * -2.885390081777927f); // -2/ln(2)
  float t  = (1.0f - e) * __builtin_amdgcn_rcpf(1.0f + e);
  return x < 0.0f ? -t : t;
}

// Y[row][j] = sum_k X[row][k] * W[k][j] + bias[j]   (K = 256, N = 512)
__global__ __launch_bounds__(256) void proj_kernel(
    const float* __restrict__ X, const float* __restrict__ W,
    const float* __restrict__ bias, float* __restrict__ Y) {
  const int row = blockIdx.x >> 1;
  const int j   = ((blockIdx.x & 1) << 8) + threadIdx.x;
  const float* xr = X + row * ENC_D;
  float acc = bias[j];
#pragma unroll 8
  for (int k = 0; k < ENC_D; ++k)
    acc = fmaf(xr[k], W[k * JD + j], acc);
  Y[row * JD + j] = acc;
}

// WoutT[v][j] = (f16) W_out[j][v]  -> column-major f16 for contiguous B-fragments
__global__ __launch_bounds__(256) void wout_transpose_kernel(
    const float* __restrict__ Wout, half_t* __restrict__ WoutT) {
  int idx = blockIdx.x * 256 + threadIdx.x;          // 512*1024 elements
  int v = idx & (VOC - 1);
  int j = idx >> 10;
  WoutT[v * JD + j] = (half_t)Wout[j * VOC + v];
}

union AF { v16h h; uint4 q[2]; };

__global__ __launch_bounds__(256, 2) void joint_wmma_kernel(
    const float*  __restrict__ enc_proj,   // [B*T][JD] f32
    const float*  __restrict__ pred_proj,  // [B*U][JD] f32
    const half_t* __restrict__ WoutT,      // [VOC][JD] f16 (col-major B)
    const float*  __restrict__ b_out,      // [VOC]
    float*        __restrict__ out) {      // [B][T][U][VOC]
  // Joint tile: 64 rows (u) x 512 (j) f16, 1024 B/row, 16-B blocks XOR-swizzled by (u&15)
  __shared__ __align__(16) unsigned char smJ[U_ * JD * 2];        // 64 KB
  // B tile double buffer: 128 cols x 128 K f16, 256 B/col, 16-B blocks XOR-swizzled by (col&15)
  __shared__ __align__(16) unsigned char smB[2][128 * 128 * 2];   // 2 x 32 KB

  const int t   = blockIdx.y;
  const int b   = blockIdx.z;
  const int nb  = blockIdx.x * 128;
  const int tid = threadIdx.x;

  // ---- B slice staging helpers (slice s = K range [s*128, s*128+128)) ----
  uint4 rb[8];
  auto loadSlice = [&](int s) {
#pragma unroll
    for (int i = 0; i < 8; ++i) {
      int c2  = tid + (i << 8);           // 2048 16-B chunks per slice
      int col = c2 >> 4;                  // 0..127
      int blk = c2 & 15;                  // 16-B block within col's 256-B row
      rb[i] = *(const uint4*)(WoutT + (size_t)(nb + col) * JD + (s << 7) + (blk << 3));
    }
  };
  auto storeSlice = [&](int p) {
#pragma unroll
    for (int i = 0; i < 8; ++i) {
      int c2  = tid + (i << 8);
      int col = c2 >> 4;
      int blk = c2 & 15;
      *(uint4*)(&smB[p][(col << 8) + ((blk ^ (col & 15)) << 4)]) = rb[i];
    }
  };

  loadSlice(0);   // global loads for slice 0 in flight during tanh staging

  // ---- stage joint tile: f16( tanh(enc[b,t,:] + pred[b,u,:]) ) ----
  const float* er = enc_proj  + (size_t)(b * T_ + t) * JD;
  const float* pr = pred_proj + (size_t)(b * U_) * JD;
  for (int c = tid; c < U_ * (JD / 8); c += 256) {  // 16-byte chunks
    int u   = c >> 6;
    int blk = c & 63;
    int j0  = blk << 3;
    float4 e0 = *(const float4*)(er + j0);
    float4 e1 = *(const float4*)(er + j0 + 4);
    const float* prr = pr + u * JD + j0;
    float4 p0 = *(const float4*)(prr);
    float4 p1 = *(const float4*)(prr + 4);
    union { half_t h[8]; uint4 q; } pk;
    pk.h[0] = (half_t)fast_tanh(e0.x + p0.x);
    pk.h[1] = (half_t)fast_tanh(e0.y + p0.y);
    pk.h[2] = (half_t)fast_tanh(e0.z + p0.z);
    pk.h[3] = (half_t)fast_tanh(e0.w + p0.w);
    pk.h[4] = (half_t)fast_tanh(e1.x + p1.x);
    pk.h[5] = (half_t)fast_tanh(e1.y + p1.y);
    pk.h[6] = (half_t)fast_tanh(e1.z + p1.z);
    pk.h[7] = (half_t)fast_tanh(e1.w + p1.w);
    ((uint4*)(smJ + u * 1024))[blk ^ (u & 15)] = pk.q;
  }
  storeSlice(0);
  __syncthreads();

  const int lane = tid & 31;
  const int wave = tid >> 5;
  const int wm   = wave & 3;        // 4 waves over M (u)
  const int wn   = wave >> 2;       // 2 waves over N (vocab)
  const int ln   = lane & 15;
  const int lh   = lane >> 4;
  const int m0   = wm << 4;
  const int nw   = nb + (wn << 6);  // this wave's 64-col base (global)
  const int cw   = wn << 6;         // this wave's 64-col base within the 128-col tile

  v8f acc0 = {}, acc1 = {}, acc2 = {}, acc3 = {};

  const uint4* ap = (const uint4*)(smJ + (m0 + ln) * 1024);

#pragma unroll
  for (int s = 0; s < 4; ++s) {
    if (s < 3) loadSlice(s + 1);           // global loads overlap with WMMAs below
    const unsigned char* bb = smB[s & 1];
    const uint4* c0p = (const uint4*)(bb + ((cw      + ln) << 8));
    const uint4* c1p = (const uint4*)(bb + ((cw + 16 + ln) << 8));
    const uint4* c2p = (const uint4*)(bb + ((cw + 32 + ln) << 8));
    const uint4* c3p = (const uint4*)(bb + ((cw + 48 + ln) << 8));
#pragma unroll
    for (int kk = 0; kk < 128; kk += 32) {
      const int k0 = (s << 7) + kk;
      // A fragment (16x32 f16): V0-3 = K(k0+lh*8 .. +7), V4-7 = K(k0+16+lh*8 .. +7)
      AF a;
      int blk0 = (k0 >> 3) + lh;
      a.q[0] = ap[(blk0    ) ^ ln];
      a.q[1] = ap[(blk0 + 2) ^ ln];
      // B fragments (32x16 f16): lane = col, V0-3 = K(k0+lh*8 .. +7), V4-7 = K(k0+16+lh*8 .. +7)
      int bi = (kk >> 3) + lh;
      AF b0f, b1f, b2f, b3f;
      b0f.q[0] = c0p[(bi) ^ ln];  b0f.q[1] = c0p[(bi + 2) ^ ln];
      b1f.q[0] = c1p[(bi) ^ ln];  b1f.q[1] = c1p[(bi + 2) ^ ln];
      b2f.q[0] = c2p[(bi) ^ ln];  b2f.q[1] = c2p[(bi + 2) ^ ln];
      b3f.q[0] = c3p[(bi) ^ ln];  b3f.q[1] = c3p[(bi + 2) ^ ln];
      acc0 = __builtin_amdgcn_wmma_f32_16x16x32_f16(false, a.h, false, b0f.h, (short)0, acc0, false, false);
      acc1 = __builtin_amdgcn_wmma_f32_16x16x32_f16(false, a.h, false, b1f.h, (short)0, acc1, false, false);
      acc2 = __builtin_amdgcn_wmma_f32_16x16x32_f16(false, a.h, false, b2f.h, (short)0, acc2, false, false);
      acc3 = __builtin_amdgcn_wmma_f32_16x16x32_f16(false, a.h, false, b3f.h, (short)0, acc3, false, false);
    }
    if (s < 3) {
      storeSlice((s + 1) & 1);   // safe: all waves are past reads of this buffer
      __syncthreads();           // publish slice s+1 before anyone computes on it
    }
  }

  const float bo0 = b_out[nw + ln];
  const float bo1 = b_out[nw + 16 + ln];
  const float bo2 = b_out[nw + 32 + ln];
  const float bo3 = b_out[nw + 48 + ln];

  float* obase = out + (size_t)(b * T_ + t) * U_ * VOC;
#pragma unroll
  for (int r = 0; r < 8; ++r) {
    // C/D layout: VGPR r -> M = r (lanes 0-15) / r+8 (lanes 16-31), N = lane%16
    int u = m0 + r + (lh << 3);
    float* orow = obase + (size_t)u * VOC + ln;
    orow[nw]      = acc0[r] + bo0;
    orow[nw + 16] = acc1[r] + bo1;
    orow[nw + 32] = acc2[r] + bo2;
    orow[nw + 48] = acc3[r] + bo3;
  }
}

extern "C" void kernel_launch(void* const* d_in, const int* in_sizes, int n_in,
                              void* d_out, int out_size, void* d_ws, size_t ws_size,
                              hipStream_t stream) {
  const float* enc   = (const float*)d_in[0];
  const float* pred  = (const float*)d_in[1];
  const float* W_enc = (const float*)d_in[2];
  const float* b_enc = (const float*)d_in[3];
  const float* W_dec = (const float*)d_in[4];
  const float* b_dec = (const float*)d_in[5];
  const float* W_out = (const float*)d_in[6];
  const float* b_out = (const float*)d_in[7];
  float* out = (float*)d_out;

  char* ws = (char*)d_ws;
  float*  enc_proj  = (float*)ws;                                     // 1024*512 f32 (2 MB)
  float*  pred_proj = (float*)(ws + (size_t)1024 * 512 * 4);          // 256*512 f32 (0.5 MB)
  half_t* WoutT     = (half_t*)(ws + (size_t)1024 * 512 * 4
                                   + (size_t)256  * 512 * 4);         // 1024*512 f16 (1 MB)

  proj_kernel<<<dim3(B_ * T_ * 2), dim3(256), 0, stream>>>(enc,  W_enc, b_enc, enc_proj);
  proj_kernel<<<dim3(B_ * U_ * 2), dim3(256), 0, stream>>>(pred, W_dec, b_dec, pred_proj);
  wout_transpose_kernel<<<dim3((JD * VOC) / 256), dim3(256), 0, stream>>>(W_out, WoutT);
  joint_wmma_kernel<<<dim3(VOC / 128, T_, B_), dim3(256), 0, stream>>>(
      enc_proj, pred_proj, WoutT, b_out, out);
}
